// LeviCivitaKANLayer_54820962566346
// MI455X (gfx1250) — matrix operand order repaired and checked by reading, not compiled, not run on hardware
//
#include <hip/hip_runtime.h>

typedef __attribute__((ext_vector_type(16))) _Float16 v16h;
typedef __attribute__((ext_vector_type(8)))  _Float16 v8h;
typedef __attribute__((ext_vector_type(8)))  float    v8f;

#define IN_DIM   128
#define OUT_DIM  128
#define KOFF     16256            // 128*127
#define NSLICE   129              // 128 pairwise slices + 1 folded-w1 slice
#define LDH      136              // padded f16 row stride (halves) -> no bank conflicts
#define LDY      132              // padded f32 row stride for LN staging
#define WSLICE_B (IN_DIM * LDH * 2)         // 34816 bytes per staged weight slice
#define XS_OFF   0
#define WB_OFF   (IN_DIM * LDH * 2)         // 34816
#define SMEM_B   (WB_OFF + 2 * WSLICE_B)    // 104448 bytes LDS in GEMM kernel

#define WP_BYTES   ((size_t)NSLICE * IN_DIM * OUT_DIM * 2)   // 4,227,072
#define BIAS_WOFF  WP_BYTES                                   // 512 B
#define PBUF_WOFF  (WP_BYTES + 512)

// ---------------------------------------------------------------------------
// Kernel 1: scatter lc_w into dense per-i slices Wp[i][o][j] (f16), diag = 0.
// Slice i==128 holds w1[o][j] * scale[j] (raw-x term folded onto xs basis).
// ---------------------------------------------------------------------------
__global__ __launch_bounds__(256) void pack_w_kernel(
    const float* __restrict__ lc_w, const float* __restrict__ w1,
    const float* __restrict__ scale, _Float16* __restrict__ Wp) {
  int idx = blockIdx.x * 256 + threadIdx.x;
  if (idx >= NSLICE * IN_DIM * OUT_DIM) return;
  int i = idx >> 14;            // / 16384
  int rem = idx & 16383;
  int o = rem >> 7;
  int j = rem & 127;
  float v;
  if (i == IN_DIM) {
    v = w1[o * IN_DIM + j] * scale[j];
  } else if (j == i) {
    v = 0.0f;
  } else {
    v = lc_w[o * KOFF + i * 127 + (j > i ? j - 1 : j)];
  }
  Wp[idx] = (_Float16)v;
}

// Folded bias: bias[o] = sum_j trans[j] * w1[o,j]  (raw-x vs xs basis shift).
__global__ __launch_bounds__(128) void bias_kernel(
    const float* __restrict__ trans, const float* __restrict__ w1,
    float* __restrict__ bias) {
  int o = threadIdx.x;
  float s = 0.0f;
#pragma unroll 4
  for (int j = 0; j < IN_DIM; ++j) s += trans[j] * w1[o * IN_DIM + j];
  bias[o] = s;
}

// Issue 8 async 16B memory->LDS copies for one 32KB weight slice.
// GVS mode: SGPR pair = Wp base, VGPR = byte offset, VGPR = LDS byte address.
// Tracked by ASYNCcnt; zero VGPR data footprint.
__device__ __forceinline__ void stage_slice_async(
    unsigned long long wp_base, int slice, unsigned lds_base, int tid) {
#pragma unroll
  for (int s = 0; s < 8; ++s) {
    int hb = (s * 256 + tid) * 8;                 // starting half index in slice
    unsigned lds  = lds_base + (unsigned)(((hb >> 7) * LDH + (hb & 127)) * 2);
    unsigned goff = (unsigned)((slice * (IN_DIM * OUT_DIM) + hb) * 2);
    asm volatile("global_load_async_to_lds_b128 %0, %1, %2 offset:0"
                 :
                 : "v"(lds), "v"(goff), "s"(wp_base)
                 : "memory");
  }
}

// ---------------------------------------------------------------------------
// Kernel 2: xs-outer-product GEMM (WMMA f16->f32), K-split over blockIdx.y.
// Block = 256 threads = 8 waves; tile = 128 batch rows x 128 outputs.
// Each (row-group, k-group) block reads a disjoint range of weight slices and
// writes its partial sums to pbuf[kg] -- no atomics, fully deterministic.
// ---------------------------------------------------------------------------
__global__ __launch_bounds__(256, 1) void lckan_gemm_kernel(
    const float* __restrict__ x, const float* __restrict__ scale,
    const float* __restrict__ trans, const _Float16* __restrict__ Wp,
    float* __restrict__ pbuf, int Btot) {
  __shared__ __align__(16) char smem[SMEM_B];
  _Float16* xs = (_Float16*)(smem + XS_OFF);

  const int tid  = threadIdx.x;
  const int wave = tid >> 5;
  const int lane = tid & 31;
  const int ml   = lane & 15;        // fragment row / column within tile
  const int hi   = lane >> 4;        // lane half
  const int klo  = hi ? 8 : 0;       // A-fragment K base within 32-chunk
  const int kkb  = hi ? 8 : 0;       // B-fragment K base within 32-chunk
  const int rowbase = blockIdx.x * 128;

  // K-range for this block: slices [kbeg, kbeg+cnt)
  const int ks   = gridDim.y;
  const int kg   = blockIdx.y;
  const int base = NSLICE / ks;
  const int rem  = NSLICE % ks;
  const int kbeg = kg * base + (kg < rem ? kg : rem);
  const int cnt  = base + (kg < rem ? 1 : 0);

  const unsigned long long wp_base = (unsigned long long)(uintptr_t)Wp;
  const unsigned smem_base = (unsigned)(uintptr_t)(&smem[0]);

  // Kick off async staging of our first weight slice into LDS buffer 0.
  stage_slice_async(wp_base, kbeg, smem_base + WB_OFF, tid);

  // Overlapped with async copy: normalize x tile into xs (f16).
  for (int idx = tid; idx < 128 * IN_DIM; idx += 256) {
    int r = idx >> 7, c = idx & 127;
    float v = x[(rowbase + r) * IN_DIM + c];
    xs[r * LDH + c] = (_Float16)((v - trans[c]) / scale[c]);
  }

  asm volatile("s_wait_asynccnt 0x0" ::: "memory");
  __syncthreads();

  v8f acc[8] = {};

  for (int t = 0; t < cnt; ++t) {
    const int i = kbeg + t;

    // Stage next slice into the spare buffer; runs during the WMMAs below.
    if (t + 1 < cnt) {
      stage_slice_async(wp_base, i + 1,
                        smem_base + WB_OFF + (unsigned)(((t + 1) & 1) * WSLICE_B),
                        tid);
      if (t + 2 < cnt)
        __builtin_prefetch((const char*)Wp + (size_t)(i + 2) * 32768 + tid * 64, 0, 1);
    }

    const _Float16* wb = (const _Float16*)(smem + WB_OFF + (t & 1) * WSLICE_B);
    _Float16 scal = (i < IN_DIM) ? xs[ml * LDH + i] : (_Float16)1.0f;

    // Build 4 A fragments: A[m, j] = xs[m, i] * xs[m, j] (diag zeroed via W).
    v16h af[4];
#pragma unroll
    for (int c = 0; c < 4; ++c) {
      const _Float16* ap = xs + ml * LDH + c * 32 + klo;
      v8h lo = *(const v8h*)ap;
      v8h hh = *(const v8h*)(ap + 16);
      v16h a = __builtin_shufflevector(lo, hh, 0, 1, 2, 3, 4, 5, 6, 7,
                                       8, 9, 10, 11, 12, 13, 14, 15);
      af[c] = a * scal;   // packed f16 scale by xs[m, i]
    }

#pragma unroll
    for (int n = 0; n < 8; ++n) {
      const _Float16* bp = wb + (n * 16 + ml) * LDH + kkb;
#pragma unroll
      for (int c = 0; c < 4; ++c) {
        v8h lo = *(const v8h*)(bp + c * 32);
        v8h hh = *(const v8h*)(bp + c * 32 + 16);
        v16h b = __builtin_shufflevector(lo, hh, 0, 1, 2, 3, 4, 5, 6, 7,
                                         8, 9, 10, 11, 12, 13, 14, 15);
        acc[n] = __builtin_amdgcn_wmma_f32_16x16x32_f16(
            false, af[c], false, b, (short)0, acc[n], false, false);
      }
    }

    if (t + 1 < cnt) {
      asm volatile("s_wait_asynccnt 0x0" ::: "memory");
      __syncthreads();
    }
  }

  // Write partial tile. C/D layout: M = vgpr + 8*hi, N = ml.
  float* pp = pbuf + (size_t)kg * Btot * OUT_DIM;
#pragma unroll
  for (int n = 0; n < 8; ++n)
#pragma unroll
    for (int r = 0; r < 8; ++r)
      pp[(size_t)(rowbase + wave * 16 + r + hi * 8) * OUT_DIM + n * 16 + ml] =
          acc[n][r];
}

// ---------------------------------------------------------------------------
// Kernel 3: reduce K-split partials + bias, LayerNorm, affine, write out.
// Block = 256 threads handles 64 rows.
// ---------------------------------------------------------------------------
__global__ __launch_bounds__(256) void ln_kernel(
    const float* __restrict__ pbuf, const float* __restrict__ bias,
    const float* __restrict__ gamma, const float* __restrict__ beta,
    float* __restrict__ out, int ks, int Btot) {
  __shared__ float yl[64 * LDY];
  const int tid = threadIdx.x;
  const int rowbase = blockIdx.x * 64;

  for (int idx = tid; idx < 64 * OUT_DIM; idx += 256) {
    int r = idx >> 7, c = idx & 127;
    float s = bias[c];
    for (int g = 0; g < ks; ++g)
      s += pbuf[(size_t)g * Btot * OUT_DIM + (size_t)(rowbase + r) * OUT_DIM + c];
    yl[r * LDY + c] = s;
  }
  __syncthreads();

  if (tid < 64) {
    const float* yr = yl + tid * LDY;
    float mu = 0.0f;
#pragma unroll 4
    for (int c = 0; c < OUT_DIM; ++c) mu += yr[c];
    mu *= (1.0f / OUT_DIM);
    float var = 0.0f;
#pragma unroll 4
    for (int c = 0; c < OUT_DIM; ++c) { float d = yr[c] - mu; var += d * d; }
    float rs = rsqrtf(var * (1.0f / OUT_DIM) + 1e-5f);
    float* op = out + (size_t)(rowbase + tid) * OUT_DIM;
#pragma unroll 4
    for (int c = 0; c < OUT_DIM; ++c)
      op[c] = (yr[c] - mu) * rs * gamma[c] + beta[c];
  }
}

// ---------------------------------------------------------------------------
extern "C" void kernel_launch(void* const* d_in, const int* in_sizes, int n_in,
                              void* d_out, int out_size, void* d_ws, size_t ws_size,
                              hipStream_t stream) {
  const float* x     = (const float*)d_in[0];
  const float* scale = (const float*)d_in[1];
  const float* trans = (const float*)d_in[2];
  const float* lc_w  = (const float*)d_in[3];
  const float* w1    = (const float*)d_in[4];
  const float* gamma = (const float*)d_in[5];
  const float* beta  = (const float*)d_in[6];
  float* out = (float*)d_out;

  const int B = in_sizes[0] / IN_DIM;   // 8192

  _Float16* Wp   = (_Float16*)d_ws;
  float*    bias = (float*)((char*)d_ws + BIAS_WOFF);
  float*    pbuf = (float*)((char*)d_ws + PBUF_WOFF);

  // Pick the largest K-split the workspace can hold (deterministic per run).
  const size_t per_split = (size_t)B * OUT_DIM * sizeof(float);
  int ks = 4;
  while (ks > 1 && PBUF_WOFF + (size_t)ks * per_split > ws_size) ks >>= 1;

  const int packN = NSLICE * IN_DIM * OUT_DIM;
  pack_w_kernel<<<(packN + 255) / 256, 256, 0, stream>>>(lc_w, w1, scale, Wp);
  bias_kernel<<<1, 128, 0, stream>>>(trans, w1, bias);

  lckan_gemm_kernel<<<dim3(B / 128, ks), 256, 0, stream>>>(x, scale, trans, Wp,
                                                           pbuf, B);
  ln_kernel<<<B / 64, 256, 0, stream>>>(pbuf, bias, gamma, beta, out, ks, B);
}
